// Multi_CA_48868137894266
// MI455X (gfx1250) — matrix operand order repaired and checked by reading, not compiled, not run on hardware
//
#include <hip/hip_runtime.h>
#include <hip/hip_bf16.h>
#include <cstdint>

// ---------------------------------------------------------------------------
// Types for CDNA5 WMMA (wave32). bf16 A/B fragments are 16 x bf16 = 8 VGPRs,
// f32 C/D fragments are 8 x f32 = 8 VGPRs.
// ---------------------------------------------------------------------------
typedef __attribute__((ext_vector_type(16))) __bf16 v16bf;
typedef __attribute__((ext_vector_type(8)))  __bf16 v8bf;
typedef __attribute__((ext_vector_type(8)))  float  v8f;

#define USE_ASYNC_LDS 1

__device__ inline v16bf bf_combine(v8bf lo, v8bf hi) {
    return __builtin_shufflevector(lo, hi, 0, 1, 2, 3, 4, 5, 6, 7,
                                          8, 9, 10, 11, 12, 13, 14, 15);
}

// D = A(16x32 bf16) * B(32x16 bf16) + C(16x16 f32)
__device__ inline v8f wmma_bf16(v16bf a, v16bf b, v8f c) {
    return __builtin_amdgcn_wmma_f32_16x16x32_bf16(
        /*neg_a=*/false, a, /*neg_b=*/false, b,
        /*c_mod=*/(short)0, c, /*reuse_a=*/false, /*reuse_b=*/false);
}

// A-matrix fragment (16x32 bf16, row-major source with leading dim `ld`).
// ISA layout: lane row = lane&15; lanes 0-15 hold K={0..7,16..23},
// lanes 16-31 hold K={8..15,24..31}.
__device__ inline v16bf load_a_frag(const __bf16* p, int ld) {
    const int lane = threadIdx.x & 31;
    const __bf16* q = p + (size_t)(lane & 15) * ld + ((lane >> 4) << 3);
    v8bf lo = *(const v8bf*)q;
    v8bf hi = *(const v8bf*)(q + 16);
    return bf_combine(lo, hi);
}

// B-matrix fragment (32x16 bf16) from *transposed* storage bt[n][k] with
// leading dim `ld` (each lane reads two 8-element K runs of one column).
// ISA layout: lane col = lane&15; lanes 0-15 hold K={0..7,16..23},
// lanes 16-31 hold K={8..15,24..31}.
__device__ inline v16bf load_b_frag(const __bf16* p, int ld) {
    const int lane = threadIdx.x & 31;
    const __bf16* q = p + (size_t)(lane & 15) * ld + ((lane >> 4) << 3);
    v8bf lo = *(const v8bf*)q;
    v8bf hi = *(const v8bf*)(q + 16);
    return bf_combine(lo, hi);
}

// ---------------------------------------------------------------------------
// Async global -> LDS staging (gfx1250 GLOBAL_LOAD_ASYNC_TO_LDS_*, ASYNCcnt).
// Copies 32 bytes (16 bf16): INST_OFFSET is added to BOTH the LDS and global
// address, so one address pair serves both halves.
// Generic LDS pointers carry the LDS byte address in their low 32 bits.
// ---------------------------------------------------------------------------
__device__ inline void stage32B(__bf16* dst_lds, const __bf16* src_global) {
#if USE_ASYNC_LDS
    uint32_t ldsoff = (uint32_t)(uintptr_t)dst_lds;
    unsigned long long gaddr = (unsigned long long)(uintptr_t)src_global;
    asm volatile("global_load_async_to_lds_b128 %0, %1, off\n\t"
                 "global_load_async_to_lds_b128 %0, %1, off offset:16"
                 :: "v"(ldsoff), "v"(gaddr)
                 : "memory");
#else
    *(v8bf*)dst_lds       = *(const v8bf*)src_global;
    *(v8bf*)(dst_lds + 8) = *(const v8bf*)(src_global + 8);
#endif
}

// Async ops complete in order: <=4 outstanding means the previous tile's
// 4 ops per wave have landed while the prefetched tile stays in flight.
__device__ inline void wait_async_keep4() {
#if USE_ASYNC_LDS
    asm volatile("s_wait_asynccnt 0x4" ::: "memory");
#endif
}
__device__ inline void wait_async_zero() {
#if USE_ASYNC_LDS
    asm volatile("s_wait_asynccnt 0x0" ::: "memory");
#endif
}

// ---------------------------------------------------------------------------
// fp32 -> bf16 convert (flat)
// ---------------------------------------------------------------------------
__global__ __launch_bounds__(256) void cvt_kernel(const float* __restrict__ in,
                                                  __bf16* __restrict__ out, int n) {
    int i = (blockIdx.x * 256 + threadIdx.x) * 4;
    if (i + 3 < n) {
        float4 v = *(const float4*)(in + i);
        out[i + 0] = (__bf16)v.x;
        out[i + 1] = (__bf16)v.y;
        out[i + 2] = (__bf16)v.z;
        out[i + 3] = (__bf16)v.w;
    }
}

// fp32 [K][N] -> bf16 [N][K] (transpose + convert), 32x32 tiles, 256 threads.
__global__ __launch_bounds__(256) void cvtT_kernel(const float* __restrict__ in,
                                                   __bf16* __restrict__ out,
                                                   int K, int N) {
    __shared__ float tile[32][33];
    const int k0 = blockIdx.y * 32, n0 = blockIdx.x * 32;
    const int tx = threadIdx.x & 31, ty = threadIdx.x >> 5;  // 32 x 8
#pragma unroll
    for (int i = 0; i < 32; i += 8)
        tile[ty + i][tx] = in[(size_t)(k0 + ty + i) * N + n0 + tx];
    __syncthreads();
#pragma unroll
    for (int i = 0; i < 32; i += 8)
        out[(size_t)(n0 + ty + i) * K + k0 + tx] = (__bf16)tile[tx][ty + i];
}

// ---------------------------------------------------------------------------
// WMMA GEMM: C[M,N] = A[M,K](bf16,row-major) * Bt[N,K](bf16, B transposed)
// 256 threads = 8 waves (4x2 grid); workgroup tile 128(M) x 128(N); K-step 32.
// Each wave computes 32x64 (2x4 WMMA tiles). A/B tiles double-buffered in LDS
// via async DMA. All fragments are loaded before the 8-WMMA burst so the DS
// loads pipeline and the WMMAs issue back-to-back.
// mode 0: Cb = bf16(C)
// mode 1: Cb = bf16(gelu(C + bias))           (exact erf GELU)
// mode 2: Cf = C + bias + resid               (fp32)
// ---------------------------------------------------------------------------
__global__ __launch_bounds__(256)
void gemm_bf16_kernel(const __bf16* __restrict__ A, const __bf16* __restrict__ Bt,
                      int M, int N, int K,
                      const float* __restrict__ bias, const float* __restrict__ resid,
                      float* __restrict__ Cf, __bf16* __restrict__ Cb, int mode) {
    __shared__ __bf16 As[2][128 * 40];   // 128 rows x 32 cols (+8 pad), x2 buffers
    __shared__ __bf16 Bs[2][128 * 40];

    const int tid  = threadIdx.x;
    const int wave = tid >> 5;
    const int lane = tid & 31;
    const int mw = (wave & 3) * 32;      // wave row offset in tile
    const int nw = (wave >> 2) * 64;     // wave col offset in tile
    const int n0 = blockIdx.x * 128;
    const int m0 = blockIdx.y * 128;

    const __bf16* Ablk = A + (size_t)m0 * K;
    const __bf16* Bblk = Bt + (size_t)n0 * K;

    // staging map: 256 threads x 32B = one 128x32 bf16 tile per matrix
    const int srow = tid >> 1;           // 0..127
    const int sseg = (tid & 1) << 4;     // 0 or 16

    v8f acc[2][4];
#pragma unroll
    for (int u = 0; u < 2; ++u)
#pragma unroll
        for (int t = 0; t < 4; ++t) acc[u][t] = v8f{};

    // prologue: issue tile 0
    stage32B(&As[0][srow * 40 + sseg], Ablk + (size_t)srow * K + sseg);
    stage32B(&Bs[0][srow * 40 + sseg], Bblk + (size_t)srow * K + sseg);

    const int nk = K >> 5;
    for (int i = 0; i < nk; ++i) {
        const int cur = i & 1;
        if (i + 1 < nk) {   // issue tile i+1 into the other buffer
            const int k = (i + 1) << 5;
            stage32B(&As[cur ^ 1][srow * 40 + sseg], Ablk + (size_t)srow * K + k + sseg);
            stage32B(&Bs[cur ^ 1][srow * 40 + sseg], Bblk + (size_t)srow * K + k + sseg);
            wait_async_keep4();
        } else {
            wait_async_zero();
        }
        __syncthreads();

        // load all fragments first, then burst the 8 WMMAs
        v16bf a0 = load_a_frag(&As[cur][(mw +  0) * 40], 40);
        v16bf a1 = load_a_frag(&As[cur][(mw + 16) * 40], 40);
        v16bf bfrag[4];
#pragma unroll
        for (int t = 0; t < 4; ++t)
            bfrag[t] = load_b_frag(&Bs[cur][(nw + t * 16) * 40], 40);
#pragma unroll
        for (int t = 0; t < 4; ++t) {
            acc[0][t] = wmma_bf16(a0, bfrag[t], acc[0][t]);
            acc[1][t] = wmma_bf16(a1, bfrag[t], acc[1][t]);
        }
        __syncthreads();            // all reads done before buffer reuse
    }

    const int hi = lane >> 4, nc = lane & 15;
#pragma unroll
    for (int u = 0; u < 2; ++u)
#pragma unroll
        for (int t = 0; t < 4; ++t) {
            const int col = n0 + nw + t * 16 + nc;
#pragma unroll
            for (int r = 0; r < 8; ++r) {
                const int row = m0 + mw + u * 16 + r + hi * 8;
                const size_t idx = (size_t)row * N + col;
                float v = acc[u][t][r];
                if (mode == 0) {
                    Cb[idx] = (__bf16)v;
                } else if (mode == 1) {
                    v += bias[col];
                    v = 0.5f * v * (1.0f + erff(v * 0.70710678118654752f));
                    Cb[idx] = (__bf16)v;
                } else {
                    v += bias[col] + resid[idx];
                    Cf[idx] = v;
                }
            }
        }
}

// ---------------------------------------------------------------------------
// 3-stream flash attention. Buffers are [B*N, H*64] bf16 (projection outputs).
// grid = (B*H, N/64); block = 128 (4 waves, 16 query rows each).
// K/V tiles double-buffered in LDS via async DMA; V cooperatively transposed
// in LDS so P*V B-fragments are contiguous ds_load_b128s.
// ---------------------------------------------------------------------------
__global__ __launch_bounds__(128)
void attn_kernel(const __bf16* __restrict__ Q,
                 const __bf16* __restrict__ K1, const __bf16* __restrict__ V1,
                 const __bf16* __restrict__ K2, const __bf16* __restrict__ V2,
                 const __bf16* __restrict__ K3, const __bf16* __restrict__ V3,
                 float* __restrict__ Out) {
    __shared__ __bf16 Ks[2][32][72];    // 32 keys x 64 dims (+8 pad), x2
    __shared__ __bf16 Vs[2][32][72];
    __shared__ __bf16 Vt[64][40];       // transposed V: 64 dims x 32 keys (+8)
    __shared__ __bf16 Ps[4][16][40];    // per-wave P staging (16 x 32 +8 pad)

    const int tid  = threadIdx.x;
    const int wave = tid >> 5;
    const int lane = tid & 31;
    const int bh = blockIdx.x;          // 0..63
    const int b = bh >> 4, h = bh & 15;
    const int qrow0 = blockIdx.y * 64 + wave * 16;
    const int rowbase = b * 1024;
    const int colbase = h * 64;
    const int ldg = 1024;
    const float scale = 0.125f;         // 64^-0.5

    const __bf16* qp = Q + (size_t)(rowbase + qrow0) * ldg + colbase;
    const v16bf qf0 = load_a_frag(qp, ldg);
    const v16bf qf1 = load_a_frag(qp + 32, ldg);

    const __bf16* Kp[3] = {K1, K2, K3};
    const __bf16* Vp[3] = {V1, V2, V3};

    // staging map: 128 threads x 32B = one 32x64 bf16 tile per matrix
    const int srow = tid >> 2;
    const int sseg = (tid & 3) << 4;
    auto issue_tile = [&](int tau, int buf) {
        const int s = tau >> 5;
        const int j = (tau & 31) << 5;
        const size_t goff = (size_t)(rowbase + j + srow) * ldg + colbase + sseg;
        stage32B(&Ks[buf][srow][sseg], Kp[s] + goff);
        stage32B(&Vs[buf][srow][sseg], Vp[s] + goff);
    };

    float oacc[4][8];
#pragma unroll
    for (int t = 0; t < 4; ++t)
#pragma unroll
        for (int r = 0; r < 8; ++r) oacc[t][r] = 0.f;

    const int hi = lane >> 4, nc = lane & 15;

    v8f o[4];
    float mi[8], li[8];

    issue_tile(0, 0);
    for (int tau = 0; tau < 96; ++tau) {
        const int cur = tau & 1;
        if ((tau & 31) == 0) {  // new stream: reset online-softmax state
#pragma unroll
            for (int t = 0; t < 4; ++t) o[t] = v8f{};
#pragma unroll
            for (int r = 0; r < 8; ++r) { mi[r] = -1e30f; li[r] = 0.f; }
        }
        if (tau + 1 < 96) {
            issue_tile(tau + 1, cur ^ 1);
            wait_async_keep4();
        } else {
            wait_async_zero();
        }
        __syncthreads();        // tile data landed for all waves

        // Cooperative transpose Vs[cur](32x64) -> Vt(64x32)
        {
            v8bf t0 = *(const v8bf*)&Vs[cur][srow][sseg];
            v8bf t1 = *(const v8bf*)&Vs[cur][srow][sseg + 8];
#pragma unroll
            for (int e = 0; e < 8; ++e) {
                Vt[sseg + e][srow]     = t0[e];
                Vt[sseg + 8 + e][srow] = t1[e];
            }
        }

        // Scores: S[16 x 32] = Q[16 x 64] * K^T, via 4 WMMAs
        v16bf kf0 = load_b_frag(&Ks[cur][0][0], 72);
        v16bf kf1 = load_b_frag(&Ks[cur][0][32], 72);
        v16bf kf2 = load_b_frag(&Ks[cur][16][0], 72);
        v16bf kf3 = load_b_frag(&Ks[cur][16][32], 72);
        v8f s0 = v8f{}, s1 = v8f{};
        s0 = wmma_bf16(qf0, kf0, s0);
        s0 = wmma_bf16(qf1, kf1, s0);
        s1 = wmma_bf16(qf0, kf2, s1);
        s1 = wmma_bf16(qf1, kf3, s1);

        __syncthreads();        // Vt ready

        // Online softmax over this 32-key block (row = r + hi*8, 16-lane halves)
        float p0[8], p1[8];
#pragma unroll
        for (int r = 0; r < 8; ++r) {
            float a0 = s0[r] * scale, a1 = s1[r] * scale;
            float mx = fmaxf(a0, a1);
#pragma unroll
            for (int off = 8; off > 0; off >>= 1)
                mx = fmaxf(mx, __shfl_xor(mx, off, 16));
            const float mnew = fmaxf(mi[r], mx);
            const float corr = __expf(mi[r] - mnew);
            const float e0 = __expf(a0 - mnew);
            const float e1 = __expf(a1 - mnew);
            float rs = e0 + e1;
#pragma unroll
            for (int off = 8; off > 0; off >>= 1)
                rs += __shfl_xor(rs, off, 16);
            li[r] = li[r] * corr + rs;
            mi[r] = mnew;
            p0[r] = e0; p1[r] = e1;
#pragma unroll
            for (int t = 0; t < 4; ++t) o[t][r] *= corr;
        }

        // Stage P (C layout -> A layout via per-wave LDS)
#pragma unroll
        for (int r = 0; r < 8; ++r) {
            Ps[wave][r + hi * 8][nc]      = (__bf16)p0[r];
            Ps[wave][r + hi * 8][16 + nc] = (__bf16)p1[r];
        }
        const v16bf pf = load_a_frag(&Ps[wave][0][0], 40);

        // O += P[16 x 32] * V[32 x 64], via 4 WMMAs (contiguous B frags from Vt)
        v16bf vf[4];
#pragma unroll
        for (int t = 0; t < 4; ++t) vf[t] = load_b_frag(&Vt[t * 16][0], 40);
#pragma unroll
        for (int t = 0; t < 4; ++t) o[t] = wmma_bf16(pf, vf[t], o[t]);

        if ((tau & 31) == 31) {  // end of stream: fold normalized result
#pragma unroll
            for (int t = 0; t < 4; ++t)
#pragma unroll
                for (int r = 0; r < 8; ++r) oacc[t][r] += o[t][r] / li[r];
        }
        __syncthreads();        // all reads done before buffer/Vt reuse
    }

#pragma unroll
    for (int t = 0; t < 4; ++t)
#pragma unroll
        for (int r = 0; r < 8; ++r) {
            const int qr = qrow0 + r + hi * 8;
            Out[(size_t)(rowbase + qr) * 1024 + colbase + t * 16 + nc] = oacc[t][r];
        }
}

// ---------------------------------------------------------------------------
// Fused residual + LayerNorm over rows of 1024. x = a (+ b), y = LN(x)*g + be.
// Optionally also writes a bf16 copy. grid = rows, block = 256.
// ---------------------------------------------------------------------------
__global__ __launch_bounds__(256)
void ln_kernel(const float* __restrict__ a, const float* __restrict__ b,
               const float* __restrict__ g, const float* __restrict__ be,
               float* __restrict__ outf, __bf16* __restrict__ outb) {
    __shared__ float red[8];
    const int row = blockIdx.x, tid = threadIdx.x;
    const size_t base = (size_t)row * 1024;

    float x[4];
    float s = 0.f;
#pragma unroll
    for (int i = 0; i < 4; ++i) {
        const int c = tid + i * 256;
        float v = a[base + c];
        if (b) v += b[base + c];
        x[i] = v;
        s += v;
    }
#pragma unroll
    for (int off = 16; off > 0; off >>= 1) s += __shfl_xor(s, off, 32);
    if ((tid & 31) == 0) red[tid >> 5] = s;
    __syncthreads();
    s = 0.f;
#pragma unroll
    for (int i = 0; i < 8; ++i) s += red[i];
    const float mu = s * (1.0f / 1024.0f);
    __syncthreads();

    float vs = 0.f;
#pragma unroll
    for (int i = 0; i < 4; ++i) {
        const float d = x[i] - mu;
        vs += d * d;
    }
#pragma unroll
    for (int off = 16; off > 0; off >>= 1) vs += __shfl_xor(vs, off, 32);
    if ((tid & 31) == 0) red[tid >> 5] = vs;
    __syncthreads();
    vs = 0.f;
#pragma unroll
    for (int i = 0; i < 8; ++i) vs += red[i];
    const float rstd = rsqrtf(vs * (1.0f / 1024.0f) + 1e-5f);

#pragma unroll
    for (int i = 0; i < 4; ++i) {
        const int c = tid + i * 256;
        const float y = (x[i] - mu) * rstd * g[c] + be[c];
        outf[base + c] = y;
        if (outb) outb[base + c] = (__bf16)y;
    }
}

// ---------------------------------------------------------------------------
// Host orchestration
// ---------------------------------------------------------------------------
extern "C" void kernel_launch(void* const* d_in, const int* in_sizes, int n_in,
                              void* d_out, int out_size, void* d_ws, size_t ws_size,
                              hipStream_t stream) {
    (void)in_sizes; (void)n_in; (void)out_size; (void)ws_size;
    const int M = 4096;   // B*N
    const int D = 1024;   // DIM
    const int F = 4096;   // FF

    const float* query = (const float*)d_in[0];
    const float* h_t   = (const float*)d_in[1];
    const float* h_a   = (const float*)d_in[2];
    const float* h_v   = (const float*)d_in[3];
    const float* Wq    = (const float*)d_in[4];
    const float* Wkt   = (const float*)d_in[5];
    const float* Wka   = (const float*)d_in[6];
    const float* Wkv   = (const float*)d_in[7];
    const float* Wvt   = (const float*)d_in[8];
    const float* Wva   = (const float*)d_in[9];
    const float* Wvv   = (const float*)d_in[10];
    const float* W1    = (const float*)d_in[11];
    const float* b1    = (const float*)d_in[12];
    const float* W2    = (const float*)d_in[13];
    const float* b2    = (const float*)d_in[14];
    const float* g1    = (const float*)d_in[15];
    const float* be1   = (const float*)d_in[16];
    const float* g2    = (const float*)d_in[17];
    const float* be2   = (const float*)d_in[18];
    float* out = (float*)d_out;

    char* w = (char*)d_ws;
    size_t off = 0;
    auto alloc = [&](size_t bytes) -> char* {
        char* p = w + off;
        off += (bytes + 255) & ~(size_t)255;
        return p;
    };

    __bf16* bq = (__bf16*)alloc((size_t)M * D * 2);
    __bf16* bt = (__bf16*)alloc((size_t)M * D * 2);
    __bf16* ba = (__bf16*)alloc((size_t)M * D * 2);
    __bf16* bv = (__bf16*)alloc((size_t)M * D * 2);
    __bf16* wqT  = (__bf16*)alloc((size_t)D * D * 2);
    __bf16* wktT = (__bf16*)alloc((size_t)D * D * 2);
    __bf16* wkaT = (__bf16*)alloc((size_t)D * D * 2);
    __bf16* wkvT = (__bf16*)alloc((size_t)D * D * 2);
    __bf16* wvtT = (__bf16*)alloc((size_t)D * D * 2);
    __bf16* wvaT = (__bf16*)alloc((size_t)D * D * 2);
    __bf16* wvvT = (__bf16*)alloc((size_t)D * D * 2);
    __bf16* w1T  = (__bf16*)alloc((size_t)F * D * 2);   // [F][D]
    __bf16* w2T  = (__bf16*)alloc((size_t)D * F * 2);   // [D][F]
    __bf16* pq  = (__bf16*)alloc((size_t)M * D * 2);
    __bf16* pkt = (__bf16*)alloc((size_t)M * D * 2);
    __bf16* pka = (__bf16*)alloc((size_t)M * D * 2);
    __bf16* pkv = (__bf16*)alloc((size_t)M * D * 2);
    __bf16* pvt = (__bf16*)alloc((size_t)M * D * 2);
    __bf16* pva = (__bf16*)alloc((size_t)M * D * 2);
    __bf16* pvv = (__bf16*)alloc((size_t)M * D * 2);
    float*  attnf = (float*)alloc((size_t)M * D * 4);
    float*  x1f   = (float*)alloc((size_t)M * D * 4);
    __bf16* x1b   = (__bf16*)alloc((size_t)M * D * 2);
    __bf16* ffh   = (__bf16*)alloc((size_t)M * F * 2);
    float*  yf    = (float*)alloc((size_t)M * D * 4);

    // 1) converts
    cvt_kernel<<<(M * D) / 1024, 256, 0, stream>>>(query, bq, M * D);
    cvt_kernel<<<(M * D) / 1024, 256, 0, stream>>>(h_t, bt, M * D);
    cvt_kernel<<<(M * D) / 1024, 256, 0, stream>>>(h_a, ba, M * D);
    cvt_kernel<<<(M * D) / 1024, 256, 0, stream>>>(h_v, bv, M * D);
    cvtT_kernel<<<dim3(D / 32, D / 32), 256, 0, stream>>>(Wq, wqT, D, D);
    cvtT_kernel<<<dim3(D / 32, D / 32), 256, 0, stream>>>(Wkt, wktT, D, D);
    cvtT_kernel<<<dim3(D / 32, D / 32), 256, 0, stream>>>(Wka, wkaT, D, D);
    cvtT_kernel<<<dim3(D / 32, D / 32), 256, 0, stream>>>(Wkv, wkvT, D, D);
    cvtT_kernel<<<dim3(D / 32, D / 32), 256, 0, stream>>>(Wvt, wvtT, D, D);
    cvtT_kernel<<<dim3(D / 32, D / 32), 256, 0, stream>>>(Wva, wvaT, D, D);
    cvtT_kernel<<<dim3(D / 32, D / 32), 256, 0, stream>>>(Wvv, wvvT, D, D);
    cvtT_kernel<<<dim3(F / 32, D / 32), 256, 0, stream>>>(W1, w1T, D, F);
    cvtT_kernel<<<dim3(D / 32, F / 32), 256, 0, stream>>>(W2, w2T, F, D);

    // 2) projections (WMMA, bf16 out): tile 128x128
    dim3 gproj(D / 128, M / 128);
    gemm_bf16_kernel<<<gproj, 256, 0, stream>>>(bq, wqT,  M, D, D, nullptr, nullptr, nullptr, pq, 0);
    gemm_bf16_kernel<<<gproj, 256, 0, stream>>>(bt, wktT, M, D, D, nullptr, nullptr, nullptr, pkt, 0);
    gemm_bf16_kernel<<<gproj, 256, 0, stream>>>(ba, wkaT, M, D, D, nullptr, nullptr, nullptr, pka, 0);
    gemm_bf16_kernel<<<gproj, 256, 0, stream>>>(bv, wkvT, M, D, D, nullptr, nullptr, nullptr, pkv, 0);
    gemm_bf16_kernel<<<gproj, 256, 0, stream>>>(bt, wvtT, M, D, D, nullptr, nullptr, nullptr, pvt, 0);
    gemm_bf16_kernel<<<gproj, 256, 0, stream>>>(ba, wvaT, M, D, D, nullptr, nullptr, nullptr, pva, 0);
    gemm_bf16_kernel<<<gproj, 256, 0, stream>>>(bv, wvvT, M, D, D, nullptr, nullptr, nullptr, pvv, 0);

    // 3) 3-stream flash attention (WMMA + async K/V staging)
    attn_kernel<<<dim3(64, 16), 128, 0, stream>>>(pq, pkt, pvt, pka, pva, pkv, pvv, attnf);

    // 4) LN1: x1 = LN(query + attn)
    ln_kernel<<<M, 256, 0, stream>>>(query, attnf, g1, be1, x1f, x1b);

    // 5) FFN1: gelu(x1 @ W1 + b1) -> bf16
    gemm_bf16_kernel<<<dim3(F / 128, M / 128), 256, 0, stream>>>(
        x1b, w1T, M, F, D, b1, nullptr, nullptr, ffh, 1);

    // 6) FFN2: x1 + (ffh @ W2 + b2) -> fp32
    gemm_bf16_kernel<<<dim3(D / 128, M / 128), 256, 0, stream>>>(
        ffh, w2T, M, D, F, b2, x1f, yf, nullptr, 2);

    // 7) LN2 -> output
    ln_kernel<<<M, 256, 0, stream>>>(yf, nullptr, g2, be2, out, nullptr);
}
